// DecoderGenerator_81020263072425
// MI455X (gfx1250) — compile-verified, matrix operand order checked by reading, not run on hardware
//
#include <hip/hip_runtime.h>
#include <hip/hip_bf16.h>
#include <math.h>
#include <stdint.h>

// Problem constants (from the reference)
#define B_    16
#define T_    128
#define V_    32000
#define EMB_  512
#define H_    512
#define L_    128      // encoder length == T
#define KFC   1024     // 2*H, fc reduction dim
#define NSLAB 8        // column slabs for the fc/LSE pass

typedef __attribute__((ext_vector_type(16))) __bf16 v16bf;
typedef __attribute__((ext_vector_type(8)))  __bf16 v8bf;
typedef __attribute__((ext_vector_type(8)))  float  v8f;
typedef __attribute__((ext_vector_type(4)))  unsigned int u32x4;
typedef __attribute__((ext_vector_type(8)))  int i32x8;
typedef __attribute__((ext_vector_type(4)))  int i32x4;

// ---------------------------------------------------------------------------
// WMMA fragment loaders (CDNA5 wave32 16x16x32 bf16 layout)
// A 16x32 (MxK): lane L -> row L%16; elem j -> K = j + (j>=8?8:0) + (L>=16?8:0)
// B 32x16 (KxN): lane L -> col L%16; same K mapping (one column per lane)
// C/D 16x16 f32: vgpr i -> row i + 8*(lane>=16); col = lane%16
// ---------------------------------------------------------------------------
__device__ __forceinline__ v16bf ld_frag16(const __bf16* p) {
  v8bf lo = *(const v8bf*)(p);        // K = kb+0..7   (16B aligned in all uses)
  v8bf hi = *(const v8bf*)(p + 16);   // K = kb+16..23
  v16bf r;
#pragma unroll
  for (int j = 0; j < 8; ++j) { r[j] = lo[j]; r[j + 8] = hi[j]; }
  return r;
}

__device__ __forceinline__ v16bf load_a(const __bf16* A, int lda, int row0, int k0, int lane) {
  int r  = row0 + (lane & 15);
  int kb = k0 + ((lane >> 4) << 3);
  return ld_frag16(A + (long)r * lda + kb);
}

// B stored transposed: Bt is N x K row-major, Bmat[k][n] = Bt[n*ldb + k]
__device__ __forceinline__ v16bf load_bt(const __bf16* Bt, int ldb, int n0, int k0, int lane) {
  int c  = n0 + (lane & 15);
  int kb = k0 + ((lane >> 4) << 3);
  return ld_frag16(Bt + (long)c * ldb + kb);
}

// ---------------------------------------------------------------------------
// Generic WMMA GEMM: C[M,N] = A[M,K](bf16) * Bt[N,K]^T + (bias0+bias1)[N] + Cadd
// One wave per 16x16 output tile. grid = (N/16, M/16), block = 32.
// ---------------------------------------------------------------------------
__global__ __launch_bounds__(32)
void gemm_bf16_kernel(const __bf16* __restrict__ A, int lda,
                      const __bf16* __restrict__ Bt, int ldb,
                      const float* __restrict__ Cadd, int ldca,
                      const float* __restrict__ bias0, const float* __restrict__ bias1,
                      float* __restrict__ C, int ldc, int K) {
  int n0   = blockIdx.x * 16;
  int row0 = blockIdx.y * 16;
  int lane = threadIdx.x & 31;
  v8f acc = {0.f, 0.f, 0.f, 0.f, 0.f, 0.f, 0.f, 0.f};
  for (int k0 = 0; k0 < K; k0 += 32) {
    v16bf a = load_a(A, lda, row0, k0, lane);
    v16bf b = load_bt(Bt, ldb, n0, k0, lane);
    acc = __builtin_amdgcn_wmma_f32_16x16x32_bf16(false, a, false, b, (short)0, acc, false, false);
  }
  int col   = n0 + (lane & 15);
  int rbase = row0 + ((lane >> 4) << 3);
  float bb = 0.f;
  if (bias0) bb += bias0[col];
  if (bias1) bb += bias1[col];
#pragma unroll
  for (int i = 0; i < 8; ++i) {
    float v = acc[i] + bb;
    if (Cadd) v += Cadd[(long)(rbase + i) * ldca + col];
    C[(long)(rbase + i) * ldc + col] = v;
  }
}

// ---------------------------------------------------------------------------
// Small elementwise / setup kernels
// ---------------------------------------------------------------------------
__global__ void cast_f32_bf16(const float* __restrict__ s, __bf16* __restrict__ d, long n) {
  long i = (long)blockIdx.x * blockDim.x + threadIdx.x;
  if (i < n) d[i] = (__bf16)s[i];
}

__global__ void init_accum_kernel(float* a) { a[0] = 0.f; a[1] = 0.f; }

__global__ void init_state_kernel(const float* __restrict__ h0, const float* __restrict__ c0,
                                  __bf16* __restrict__ hs, float* __restrict__ cs) {
  int i = blockIdx.x * blockDim.x + threadIdx.x;
  if (i < 2 * B_ * H_) { hs[i] = (__bf16)h0[i]; cs[i] = c0[i]; }
}

// attn_W (H x 2H): Wh = attn_W[:, :H], We = attn_W[:, H:]
__global__ void split_attn_kernel(const float* __restrict__ attnW,
                                  __bf16* __restrict__ WhBf, __bf16* __restrict__ WeBf) {
  int i = blockIdx.x * blockDim.x + threadIdx.x;
  if (i < H_ * H_) {
    int k = i / H_, h = i % H_;
    WhBf[i] = (__bf16)attnW[k * 2 * H_ + h];
    WeBf[i] = (__bf16)attnW[k * 2 * H_ + H_ + h];
  }
}

// encBf[b][l][h] and encT[b][h][l] (transposed copy so all GEMMs use Bt loads)
__global__ void enc_cast_kernel(const float* __restrict__ enc,
                                __bf16* __restrict__ encBf, __bf16* __restrict__ encT) {
  long i = (long)blockIdx.x * blockDim.x + threadIdx.x;
  if (i < (long)B_ * L_ * H_) {
    int  h  = (int)(i % H_);
    long bl = i / H_;
    int  l  = (int)(bl % L_);
    int  b  = (int)(bl / L_);
    float v = enc[i];
    encBf[i] = (__bf16)v;
    encT[(long)(b * H_ + h) * L_ + l] = (__bf16)v;
  }
}

// Ebf[t][b][e] = (bf16) embedding[X[b][t]][e]
__global__ void embed_gather_kernel(const int* __restrict__ X, const float* __restrict__ emb,
                                    __bf16* __restrict__ E) {
  long i = (long)blockIdx.x * blockDim.x + threadIdx.x;
  if (i < (long)T_ * B_ * EMB_) {
    int  e  = (int)(i % EMB_);
    long tb = i / EMB_;
    int  b  = (int)(tb % B_);
    int  t  = (int)(tb / B_);
    int tok = X[b * (T_ + 1) + t];
    E[i] = (__bf16)emb[(long)tok * EMB_ + e];
  }
}

// LSTM pointwise: gates[B,4H] laid out [i|f|g|o]; updates c (f32) and h (bf16)
__global__ void lstm_elem_kernel(const float* __restrict__ gates, float* __restrict__ cstate,
                                 __bf16* __restrict__ hstate, __bf16* __restrict__ hall_t) {
  int idx = blockIdx.x * blockDim.x + threadIdx.x;
  if (idx >= B_ * H_) return;
  int b = idx / H_, h = idx % H_;
  const float* g = gates + b * 4 * H_;
  float gi = g[h], gf = g[H_ + h], gg = g[2 * H_ + h], go = g[3 * H_ + h];
  float c  = cstate[idx];
  float si = 1.f / (1.f + __expf(-gi));
  float sf = 1.f / (1.f + __expf(-gf));
  float so = 1.f / (1.f + __expf(-go));
  float c2 = sf * c + si * tanhf(gg);
  float h2 = so * tanhf(c2);
  cstate[idx] = c2;
  hstate[idx] = (__bf16)h2;
  if (hall_t) hall_t[idx] = (__bf16)h2;
}

// logits[t,b,l] = sum_k v_w[k] * tanh(ph[t,b,k] + pe[b,l,k])  (attn_b folded into pe)
__global__ void attn_logits_kernel(const float* __restrict__ ph, const float* __restrict__ pe,
                                   const float* __restrict__ vw, float* __restrict__ logits) {
  int wid  = (blockIdx.x * blockDim.x + threadIdx.x) >> 5;
  int lane = threadIdx.x & 31;
  if (wid >= T_ * B_ * L_) return;
  int l  = wid % L_;
  int tb = wid / L_;
  int b  = tb % B_;
  int t  = tb / B_;
  const float* pph = ph + (long)(t * B_ + b) * H_;
  const float* ppe = pe + (long)(b * L_ + l) * H_;
  float s = 0.f;
  for (int k = lane; k < H_; k += 32) s += vw[k] * tanhf(pph[k] + ppe[k]);
#pragma unroll
  for (int m = 16; m; m >>= 1) s += __shfl_xor(s, m, 32);
  if (lane == 0) logits[(long)(t * B_ + b) * L_ + l] = s;
}

// softmax over axis=1 (batch), exactly as the reference does (mask is all-false)
__global__ void attn_softmax_b_kernel(const float* __restrict__ logits,
                                      __bf16* __restrict__ attBf) {
  int idx = blockIdx.x * blockDim.x + threadIdx.x;  // t*L + l
  if (idx >= T_ * L_) return;
  int t = idx / L_, l = idx % L_;
  float m = -3.0e38f;
  for (int b = 0; b < B_; ++b) m = fmaxf(m, logits[(long)(t * B_ + b) * L_ + l]);
  float s = 0.f;
  for (int b = 0; b < B_; ++b) s += __expf(logits[(long)(t * B_ + b) * L_ + l] - m);
  float inv = 1.f / s;
  for (int b = 0; b < B_; ++b) {
    long o = (long)(t * B_ + b) * L_ + l;
    attBf[o] = (__bf16)(__expf(logits[o] - m) * inv);
  }
}

// Acat[r, 0:H]=H_all[t,b,:], Acat[r, H:2H]=weighted[t,b,:]; r = b*T + t
__global__ void build_acat_kernel(const __bf16* __restrict__ HallBf,
                                  const float* __restrict__ weighted,
                                  __bf16* __restrict__ Acat) {
  long i = (long)blockIdx.x * blockDim.x + threadIdx.x;  // r*H + h
  if (i < (long)T_ * B_ * H_) {
    int h = (int)(i % H_);
    int r = (int)(i / H_);
    int t = r % T_, b = r / T_;
    long src = (long)(t * B_ + b) * H_ + h;
    Acat[(long)r * KFC + h]      = HallBf[src];
    Acat[(long)r * KFC + H_ + h] = (__bf16)weighted[src];
  }
}

// ---------------------------------------------------------------------------
// Fused fc GEMM + online log-sum-exp (partial over a column slab of V).
// grid = (row tiles = 128, slabs = 8); block = 256 (8 waves).
// A-tile (16x1024 bf16 = 32KB) DMA'd into LDS by the Tensor Data Mover.
// ---------------------------------------------------------------------------
__global__ __launch_bounds__(256)
void fc_lse_partial_kernel(const __bf16* __restrict__ Acat, const __bf16* __restrict__ Wbf,
                           const float* __restrict__ bias,
                           float* __restrict__ pm, float* __restrict__ ps) {
  __shared__ __bf16 As[16 * KFC];   // 32 KB
  __shared__ float red_m[8][16];
  __shared__ float red_s[8][16];
  int row0 = blockIdx.x * 16;
  int slab = blockIdx.y;

#if __has_builtin(__builtin_amdgcn_tensor_load_to_lds)
  if (threadIdx.x < 32) {
    // Tensor DMA descriptor (D#): 2D tile, 16 rows x 1024 cols of 2-byte data.
    unsigned lds_addr = (unsigned)(uintptr_t)(void*)(&As[0]);       // low 32 = LDS offset
    unsigned long long ga = (unsigned long long)(uintptr_t)(Acat + (size_t)row0 * KFC);
    u32x4 g0;
    g0[0] = 1u;                                                     // count=1, user D#
    g0[1] = lds_addr;                                               // lds_addr
    g0[2] = (unsigned)(ga & 0xffffffffu);                           // global_addr[31:0]
    g0[3] = (unsigned)((ga >> 32) & 0x1ffffffu) | (2u << 30);       // addr[56:32], type=2
    i32x8 g1;
    g1[0] = 0x00010000;                                             // data_size=1 (2 bytes)
    g1[1] = (int)(KFC << 16);                                       // tensor_dim0 = 1024
    g1[2] = (int)(16 << 16);                                        // tensor_dim1 = 16
    g1[3] = (int)(KFC << 16);                                       // tile_dim0 = 1024
    g1[4] = 16;                                                     // tile_dim1 = 16
    g1[5] = KFC;                                                    // tensor_dim0_stride
    g1[6] = 0;
    g1[7] = 0;
    i32x4 gz = {0, 0, 0, 0};
#if __clang_major__ >= 23
    i32x8 gz8 = {0, 0, 0, 0, 0, 0, 0, 0};
    __builtin_amdgcn_tensor_load_to_lds(g0, g1, gz, gz, gz8, 0);
#else
    __builtin_amdgcn_tensor_load_to_lds(g0, g1, gz, gz, 0);
#endif
    __builtin_amdgcn_s_wait_tensorcnt(0);
  }
#else
  for (int i = threadIdx.x; i < 16 * KFC; i += blockDim.x)
    As[i] = Acat[(long)(row0 + i / KFC) * KFC + (i % KFC)];
#endif
  __syncthreads();

  int wave = threadIdx.x >> 5, lane = threadIdx.x & 31;
  float m[8], s[8];
#pragma unroll
  for (int i = 0; i < 8; ++i) { m[i] = -3.0e38f; s[i] = 0.f; }

  const int chunks_per_slab = (V_ / 16) / NSLAB;   // 250
  const int ch0 = slab * chunks_per_slab;
  const int ch1 = ch0 + chunks_per_slab;
  for (int ch = ch0 + wave; ch < ch1; ch += 8) {
    int n0 = ch * 16;
    v8f acc = {0.f, 0.f, 0.f, 0.f, 0.f, 0.f, 0.f, 0.f};
    for (int k0 = 0; k0 < KFC; k0 += 32) {
      v16bf a = load_a(As, KFC, 0, k0, lane);
      v16bf b = load_bt(Wbf, KFC, n0, k0, lane);
      acc = __builtin_amdgcn_wmma_f32_16x16x32_bf16(false, a, false, b, (short)0, acc, false, false);
    }
    float bb = bias[n0 + (lane & 15)];
#pragma unroll
    for (int i = 0; i < 8; ++i) {       // online max/sum per (row, this lane's cols)
      float z = acc[i] + bb;
      if (z > m[i]) { s[i] = s[i] * __expf(m[i] - z) + 1.f; m[i] = z; }
      else          { s[i] += __expf(z - m[i]); }
    }
  }
  // reduce across the 16 columns (xor d<16 stays within each 16-lane half)
#pragma unroll
  for (int d = 1; d < 16; d <<= 1) {
#pragma unroll
    for (int i = 0; i < 8; ++i) {
      float om = __shfl_xor(m[i], d, 32);
      float os = __shfl_xor(s[i], d, 32);
      if (om > m[i]) { s[i] = s[i] * __expf(m[i] - om) + os; m[i] = om; }
      else           { s[i] += os * __expf(om - m[i]); }
    }
  }
  if ((lane & 15) == 0) {               // lane0 -> rows 0..7, lane16 -> rows 8..15
    int rbase = (lane >> 4) * 8;
#pragma unroll
    for (int i = 0; i < 8; ++i) { red_m[wave][rbase + i] = m[i]; red_s[wave][rbase + i] = s[i]; }
  }
  __syncthreads();
  if (threadIdx.x < 16) {
    int r = threadIdx.x;
    float M = -3.0e38f, S = 0.f;
    for (int w = 0; w < 8; ++w) {
      float om = red_m[w][r], os = red_s[w][r];
      if (om > M) { S = S * __expf(M - om) + os; M = om; }
      else        { S += os * __expf(om - M); }
    }
    pm[slab * (T_ * B_) + row0 + r] = M;
    ps[slab * (T_ * B_) + row0 + r] = S;
  }
}

// merge slab partials -> log-sum-exp per row
__global__ void lse_combine_kernel(const float* __restrict__ pm, const float* __restrict__ ps,
                                   float* __restrict__ lse) {
  int r = blockIdx.x * blockDim.x + threadIdx.x;
  if (r >= T_ * B_) return;
  float M = -3.0e38f, S = 0.f;
  for (int sl = 0; sl < NSLAB; ++sl) {
    float om = pm[sl * (T_ * B_) + r], os = ps[sl * (T_ * B_) + r];
    if (om > M) { S = S * __expf(M - om) + os; M = om; }
    else        { S += os * __expf(om - M); }
  }
  lse[r] = M + __logf(S);
}

// exact f32 target logit + NLL accumulation; one wave per row r = b*T + t
__global__ void zy_nll_kernel(const __bf16* __restrict__ Acat, const float* __restrict__ fcW,
                              const float* __restrict__ fcb, const float* __restrict__ lse,
                              const int* __restrict__ X, float* __restrict__ accum) {
  int wid  = (blockIdx.x * blockDim.x + threadIdx.x) >> 5;
  int lane = threadIdx.x & 31;
  if (wid >= T_ * B_) return;
  int b = wid / T_, t = wid % T_;
  int y = X[b * (T_ + 1) + t + 1];
  const __bf16* a = Acat + (long)wid * KFC;
  const float*  w = fcW  + (long)y * KFC;
  float s = 0.f;
  for (int k = lane; k < KFC; k += 32) s += (float)a[k] * w[k];
#pragma unroll
  for (int m = 16; m; m >>= 1) s += __shfl_xor(s, m, 32);
  if (lane == 0 && y != 0) {            // PAD == 0
    float nll = lse[wid] - (s + fcb[y]);
    atomicAdd(&accum[0], nll);
    atomicAdd(&accum[1], 1.f);
  }
}

__global__ void finalize_kernel(const float* __restrict__ accum, float* __restrict__ out) {
  out[0] = accum[0] / accum[1];
}

// ---------------------------------------------------------------------------
// Host-side orchestration
// ---------------------------------------------------------------------------
static inline void gemm(hipStream_t st, const __bf16* A, int lda,
                        const __bf16* Bt, int ldb,
                        const float* Cadd, int ldca, const float* b0, const float* b1,
                        float* C, int ldc, int M, int N, int K) {
  dim3 g(N / 16, M / 16);
  gemm_bf16_kernel<<<g, 32, 0, st>>>(A, lda, Bt, ldb, Cadd, ldca, b0, b1, C, ldc, K);
}

static inline void cast_bf(hipStream_t st, const float* s, __bf16* d, long n) {
  cast_f32_bf16<<<(int)((n + 255) / 256), 256, 0, st>>>(s, d, n);
}

extern "C" void kernel_launch(void* const* d_in, const int* in_sizes, int n_in,
                              void* d_out, int out_size, void* d_ws, size_t ws_size,
                              hipStream_t stream) {
  const int*   X     = (const int*)  d_in[0];
  const float* enc   = (const float*)d_in[2];
  const float* emb   = (const float*)d_in[3];
  const float* Wih0  = (const float*)d_in[4];
  const float* Whh0  = (const float*)d_in[5];
  const float* bih0  = (const float*)d_in[6];
  const float* bhh0  = (const float*)d_in[7];
  const float* Wih1  = (const float*)d_in[8];
  const float* Whh1  = (const float*)d_in[9];
  const float* bih1  = (const float*)d_in[10];
  const float* bhh1  = (const float*)d_in[11];
  const float* attnW = (const float*)d_in[12];
  const float* attnb = (const float*)d_in[13];
  const float* vw    = (const float*)d_in[14];
  const float* fcW   = (const float*)d_in[15];
  const float* fcb   = (const float*)d_in[16];
  const float* h0    = (const float*)d_in[17];
  const float* c0    = (const float*)d_in[18];
  float* out = (float*)d_out;

  uint8_t* ws  = (uint8_t*)d_ws;
  size_t   off = 0;
  auto alloc = [&](size_t bytes) -> void* {
    void* p = ws + off;
    off += (bytes + 255) & ~(size_t)255;
    return p;
  };

  __bf16* Ebf     = (__bf16*)alloc((size_t)T_ * B_ * EMB_ * 2);
  __bf16* Wih0bf  = (__bf16*)alloc((size_t)4 * H_ * EMB_ * 2);
  __bf16* Whh0bf  = (__bf16*)alloc((size_t)4 * H_ * H_ * 2);
  __bf16* Wih1bf  = (__bf16*)alloc((size_t)4 * H_ * H_ * 2);
  __bf16* Whh1bf  = (__bf16*)alloc((size_t)4 * H_ * H_ * 2);
  __bf16* fcWbf   = (__bf16*)alloc((size_t)V_ * KFC * 2);
  __bf16* WhBf    = (__bf16*)alloc((size_t)H_ * H_ * 2);
  __bf16* WeBf    = (__bf16*)alloc((size_t)H_ * H_ * 2);
  __bf16* encBf   = (__bf16*)alloc((size_t)B_ * L_ * H_ * 2);
  __bf16* encT    = (__bf16*)alloc((size_t)B_ * H_ * L_ * 2);
  float*  X0      = (float*) alloc((size_t)T_ * B_ * 4 * H_ * 4);
  __bf16* hstate  = (__bf16*)alloc((size_t)2 * B_ * H_ * 2);
  float*  cstate  = (float*) alloc((size_t)2 * B_ * H_ * 4);
  float*  gates0  = (float*) alloc((size_t)B_ * 4 * H_ * 4);
  float*  gates1  = (float*) alloc((size_t)B_ * 4 * H_ * 4);
  __bf16* HallBf  = (__bf16*)alloc((size_t)T_ * B_ * H_ * 2);
  float*  ph      = (float*) alloc((size_t)T_ * B_ * H_ * 4);
  float*  pe      = (float*) alloc((size_t)B_ * L_ * H_ * 4);
  float*  logits  = (float*) alloc((size_t)T_ * B_ * L_ * 4);
  __bf16* attBf   = (__bf16*)alloc((size_t)T_ * B_ * L_ * 2);
  float*  wtd     = (float*) alloc((size_t)T_ * B_ * H_ * 4);
  __bf16* Acat    = (__bf16*)alloc((size_t)T_ * B_ * KFC * 2);
  float*  pm      = (float*) alloc((size_t)NSLAB * T_ * B_ * 4);
  float*  psum    = (float*) alloc((size_t)NSLAB * T_ * B_ * 4);
  float*  lse     = (float*) alloc((size_t)T_ * B_ * 4);
  float*  accum   = (float*) alloc(2 * 4);

  // ---- setup: accumulators, weight casts, embedding gather, state init ----
  init_accum_kernel<<<1, 1, 0, stream>>>(accum);
  cast_bf(stream, Wih0, Wih0bf, (long)4 * H_ * EMB_);
  cast_bf(stream, Whh0, Whh0bf, (long)4 * H_ * H_);
  cast_bf(stream, Wih1, Wih1bf, (long)4 * H_ * H_);
  cast_bf(stream, Whh1, Whh1bf, (long)4 * H_ * H_);
  cast_bf(stream, fcW,  fcWbf,  (long)V_ * KFC);
  enc_cast_kernel<<<(int)(((long)B_ * L_ * H_ + 255) / 256), 256, 0, stream>>>(enc, encBf, encT);
  split_attn_kernel<<<(H_ * H_ + 255) / 256, 256, 0, stream>>>(attnW, WhBf, WeBf);
  embed_gather_kernel<<<(int)(((long)T_ * B_ * EMB_ + 255) / 256), 256, 0, stream>>>(X, emb, Ebf);
  init_state_kernel<<<(2 * B_ * H_ + 255) / 256, 256, 0, stream>>>(h0, c0, hstate, cstate);

  // ---- X0 = E @ W_ih0^T + (b_ih0 + b_hh0), all timesteps at once ----
  gemm(stream, Ebf, EMB_, Wih0bf, EMB_, nullptr, 0, bih0, bhh0,
       X0, 4 * H_, T_ * B_, 4 * H_, EMB_);

  // ---- sequential 2-layer LSTM scan ----
  __bf16* hs0 = hstate;             __bf16* hs1 = hstate + B_ * H_;
  float*  cs0 = cstate;             float*  cs1 = cstate + B_ * H_;
  for (int t = 0; t < T_; ++t) {
    gemm(stream, hs0, H_, Whh0bf, H_, X0 + (size_t)t * B_ * 4 * H_, 4 * H_,
         nullptr, nullptr, gates0, 4 * H_, B_, 4 * H_, H_);
    lstm_elem_kernel<<<(B_ * H_ + 255) / 256, 256, 0, stream>>>(gates0, cs0, hs0, nullptr);
    gemm(stream, hs0, H_, Wih1bf, H_, nullptr, 0, bih1, bhh1,
         gates1, 4 * H_, B_, 4 * H_, H_);
    gemm(stream, hs1, H_, Whh1bf, H_, gates1, 4 * H_, nullptr, nullptr,
         gates1, 4 * H_, B_, 4 * H_, H_);
    lstm_elem_kernel<<<(B_ * H_ + 255) / 256, 256, 0, stream>>>(
        gates1, cs1, hs1, HallBf + (size_t)t * B_ * H_);
  }

  // ---- attention ----
  gemm(stream, HallBf, H_, WhBf, H_, nullptr, 0, nullptr, nullptr,
       ph, H_, T_ * B_, H_, H_);
  gemm(stream, encBf, H_, WeBf, H_, nullptr, 0, attnb, nullptr,
       pe, H_, B_ * L_, H_, H_);                               // attn_b folded here
  attn_logits_kernel<<<(T_ * B_ * L_) / 8, 256, 0, stream>>>(ph, pe, vw, logits);
  attn_softmax_b_kernel<<<(T_ * L_ + 255) / 256, 256, 0, stream>>>(logits, attBf);
  for (int b = 0; b < B_; ++b) {
    gemm(stream, attBf + b * L_, B_ * L_,                      // A: att[:,b,:] strided
         encT + (size_t)b * H_ * L_, L_,                       // Bt: enc[b]^T (H x L)
         nullptr, 0, nullptr, nullptr,
         wtd + b * H_, B_ * H_,                                // C: weighted[:,b,:] strided
         T_, H_, L_);
  }
  build_acat_kernel<<<(int)(((long)T_ * B_ * H_ + 255) / 256), 256, 0, stream>>>(HallBf, wtd, Acat);

  // ---- fused fc + online log-sum-exp (TDM-staged A tile), exact z_y, NLL mean ----
  {
    dim3 g(T_ * B_ / 16, NSLAB);
    fc_lse_partial_kernel<<<g, 256, 0, stream>>>(Acat, fcWbf, fcb, pm, psum);
  }
  lse_combine_kernel<<<(T_ * B_ + 255) / 256, 256, 0, stream>>>(pm, psum, lse);
  zy_nll_kernel<<<(T_ * B_) / 8, 256, 0, stream>>>(Acat, fcW, fcb, lse, X, accum);
  finalize_kernel<<<1, 1, 0, stream>>>(accum, out);

  (void)in_sizes; (void)n_in; (void)out_size; (void)ws_size;
}
